// ionBrain_14654428414750
// MI455X (gfx1250) — matrix operand order repaired and checked by reading, not compiled, not run on hardware
//
#include <hip/hip_runtime.h>
#include <stdint.h>

typedef __attribute__((ext_vector_type(16))) __bf16 v16bf;
typedef __attribute__((ext_vector_type(8)))  float  v8f;

#define Bb      4
#define Tb      8192
#define Dm      512
#define NPp     1024
#define FFd     2048
#define PATCHs  8
#define NTrows  (Bb*Tb)     // 32768
#define NProws  (Bb*NPp)    // 4096

// ------------------------------------------------ gfx1250 async-LDS helpers
// Copies 16 bytes per lane from global memory directly into LDS, tracked by
// ASYNCcnt (no VGPR data path). LDS address = low 32 bits of flat pointer
// (ISA: LDS aperture truncates to addr[31:0]).
__device__ __forceinline__ void async_ld_b128(void* lds, const void* gptr) {
    uint32_t l = (uint32_t)(uintptr_t)lds;
    asm volatile("global_load_async_to_lds_b128 %0, %1, off"
                 :: "v"(l), "v"(gptr) : "memory");
}
__device__ __forceinline__ void wait_async0() {
    asm volatile("s_wait_asynccnt 0x0" ::: "memory");
}

// ---------------------------------------------------------------- reductions
__device__ __forceinline__ float blockReduceSum(float v, float* red) {
    int tid = threadIdx.x;
    red[tid] = v; __syncthreads();
    for (int s = 128; s > 0; s >>= 1) {
        if (tid < s) red[tid] += red[tid + s];
        __syncthreads();
    }
    float r = red[0]; __syncthreads();
    return r;
}

// ------------------------------------------------------- embedding + entropy
__global__ __launch_bounds__(256)
void k_embed_entropy(const int* __restrict__ bytes,
                     const float* __restrict__ byte_embed,
                     const float* __restrict__ ent_embed,
                     const float* __restrict__ ent_w,   // (D,4)
                     const float* __restrict__ ent_b,   // (D)
                     const float* __restrict__ projW,   // (D)
                     const float* __restrict__ projB,   // (1)
                     float* __restrict__ emb,           // (B*T, D)
                     float* __restrict__ entropy) {     // (B*T)
    __shared__ float red[256];
    int bt = blockIdx.x;
    int t = bt % Tb;
    int tid = threadIdx.x;
    int tok[4];
    #pragma unroll
    for (int j = 0; j < 4; ++j) {
        int tt = t - 3 + j;
        tok[j] = (tt >= 0) ? bytes[bt - t + tt] : -1;
    }
    int tokc = bytes[bt];
    float partial = 0.f;
    for (int d = tid; d < Dm; d += 256) {
        float e = ent_b[d];
        #pragma unroll
        for (int j = 0; j < 4; ++j)
            if (tok[j] >= 0) e += ent_w[d * 4 + j] * ent_embed[(size_t)tok[j] * Dm + d];
        partial += e * projW[d];
        emb[(size_t)bt * Dm + d] = byte_embed[(size_t)tokc * Dm + d];
    }
    float s = blockReduceSum(partial, red);
    if (tid == 0) entropy[bt] = 1.f / (1.f + expf(-(s + projB[0])));
}

// ---------------------------------------------------- LNN depthwise conv k=7
__global__ __launch_bounds__(256)
void k_lnn_conv(const float* __restrict__ x,     // (B*T,D)
                const float* __restrict__ gw, const float* __restrict__ gb,
                const float* __restrict__ cw, const float* __restrict__ cb,
                float* __restrict__ gated) {     // (B*T,D)
    int bt = blockIdx.x;
    int t = bt % Tb;
    int tid = threadIdx.x;
    for (int d = tid; d < Dm; d += 256) {
        float g = gb[d], c = cb[d];
        #pragma unroll
        for (int j = 0; j < 7; ++j) {
            int tt = t - 3 + j;
            if (tt >= 0 && tt < Tb) {
                float xv = x[(size_t)(bt - t + tt) * Dm + d];
                g += gw[d * 7 + j] * xv;
                c += cw[d * 7 + j] * xv;
            }
        }
        float sg = 1.f / (1.f + expf(-g));
        gated[(size_t)bt * Dm + d] = sg * tanhf(c);
    }
}

// ------------------------------------- LNN decay scan + running-mean (fused)
__global__ void k_lnn_scan(const float* __restrict__ gated,
                           const float* __restrict__ log_decay,
                           float* __restrict__ H, float* __restrict__ I) {
    int gid = blockIdx.x * 256 + threadIdx.x;
    if (gid >= Bb * Dm) return;
    int b = gid / Dm, d = gid % Dm;
    float dec = expf(log_decay[d]);
    float h = 0.f, s = 0.f;
    size_t base = (size_t)b * Tb * Dm + d;
    for (int t = 0; t < Tb; ++t) {
        h = dec * h + gated[base + (size_t)t * Dm];
        s += h;
        H[base + (size_t)t * Dm] = h;
        I[base + (size_t)t * Dm] = s / (float)(t + 1);
    }
}

// ------------------------------------------- cumulative mean (patch level)
__global__ void k_cumsum_mean(const float* __restrict__ X, float* __restrict__ I, int Tt) {
    int gid = blockIdx.x * 256 + threadIdx.x;
    if (gid >= Bb * Dm) return;
    int b = gid / Dm, d = gid % Dm;
    float s = 0.f;
    size_t base = (size_t)b * Tt * Dm + d;
    for (int t = 0; t < Tt; ++t) {
        s += X[base + (size_t)t * Dm];
        I[base + (size_t)t * Dm] = s / (float)(t + 1);
    }
}

// ------- PID prep: gates via softmax, pack [g0*x | g1*int | g2*deriv] (bf16)
__global__ __launch_bounds__(256)
void k_pid_prep(const float* __restrict__ X, const float* __restrict__ I,
                const float* __restrict__ gW, const float* __restrict__ gB,
                __bf16* __restrict__ Acat, int Tt) {
    __shared__ float red[256];
    int row = blockIdx.x;
    int t = row % Tt;
    int tid = threadIdx.x;
    const float* x  = X + (size_t)row * Dm;
    const float* xi = I + (size_t)row * Dm;
    float p0 = 0.f, p1 = 0.f, p2 = 0.f;
    for (int d = tid; d < Dm; d += 256) {
        float v = x[d];
        p0 += v * gW[d];
        p1 += v * gW[Dm + d];
        p2 += v * gW[2 * Dm + d];
    }
    float l0 = blockReduceSum(p0, red) + gB[0];
    float l1 = blockReduceSum(p1, red) + gB[1];
    float l2 = blockReduceSum(p2, red) + gB[2];
    float m = fmaxf(l0, fmaxf(l1, l2));
    float e0 = expf(l0 - m), e1 = expf(l1 - m), e2 = expf(l2 - m);
    float inv = 1.f / (e0 + e1 + e2);
    float g0 = e0 * inv, g1 = e1 * inv, g2 = e2 * inv;
    __bf16* out = Acat + (size_t)row * (3 * Dm);
    for (int d = tid; d < Dm; d += 256) {
        float v = x[d];
        float dv = (t > 0) ? (v - x[d - Dm]) : 0.f;
        out[d]          = (__bf16)(g0 * v);
        out[Dm + d]     = (__bf16)(g1 * xi[d]);
        out[2 * Dm + d] = (__bf16)(g2 * dv);
    }
}

// --------------------------------------------------------------- weight packs
__global__ void k_pack_w3(const float* __restrict__ wp, const float* __restrict__ wi,
                          const float* __restrict__ wd, __bf16* __restrict__ out,
                          int M, int Kd) {
    size_t n = (size_t)M * 3 * Kd;
    for (size_t i = (size_t)blockIdx.x * 256 + threadIdx.x; i < n; i += (size_t)gridDim.x * 256) {
        size_t m = i / (3 * Kd);
        int k = (int)(i % (3 * Kd));
        float v = (k < Kd) ? wp[m * Kd + k]
                : (k < 2 * Kd) ? wi[m * Kd + k - Kd]
                               : wd[m * Kd + k - 2 * Kd];
        out[i] = (__bf16)v;
    }
}

__global__ void k_f32_to_bf16(const float* __restrict__ x, __bf16* __restrict__ y, size_t n) {
    for (size_t i = (size_t)blockIdx.x * 256 + threadIdx.x; i < n; i += (size_t)gridDim.x * 256)
        y[i] = (__bf16)x[i];
}

// -------------------------------------------------------------- WMMA GEMM
// C(N,M) = A(N,K) * W(M,K)^T (+bias). bf16 in, f32 accumulate.
// 256 threads = 8 waves; 128x128 C tile; BK=32 (one v_wmma_f32_16x16x32_bf16
// K-step). Double-buffered LDS filled by global_load_async_to_lds_b128
// (ASYNCcnt); next tile's fill overlaps current tile's WMMA burst.
// Pipeline order per ISA: issue-next -> compute -> s_wait_asynccnt 0 ->
// barrier (wait precedes barrier because ASYNCcnt is per-wave).
__global__ __launch_bounds__(256)
void k_gemm_bf16(const __bf16* __restrict__ A, const __bf16* __restrict__ W,
                 const float* __restrict__ bias, float* __restrict__ C,
                 int N, int K, int M) {
    __shared__ __bf16 As[2][128][32];
    __shared__ __bf16 Bs[2][128][32];
    const int tid  = threadIdx.x;
    const int lane = tid & 31;
    const int wave = tid >> 5;
    const int waveRow = wave >> 2;     // 0..1 -> 64 token rows
    const int waveCol = wave & 3;      // 0..3 -> 32 output cols
    const int rowG0 = blockIdx.y * 128;
    const int colG0 = blockIdx.x * 128;

    v8f acc[4][2];
    #pragma unroll
    for (int tb = 0; tb < 4; ++tb)
        #pragma unroll
        for (int cbI = 0; cbI < 2; ++cbI)
            acc[tb][cbI] = (v8f){0.f, 0.f, 0.f, 0.f, 0.f, 0.f, 0.f, 0.f};

    const int ldRow  = tid >> 1;
    const int ldHalf = (tid & 1) * 16;
    const __bf16* aBase = A + (size_t)(rowG0 + ldRow) * K + ldHalf;
    const __bf16* wBase = W + (size_t)(colG0 + ldRow) * K + ldHalf;

    // prologue: fill buffer 0
    async_ld_b128(&As[0][ldRow][ldHalf],     aBase);
    async_ld_b128(&As[0][ldRow][ldHalf + 8], aBase + 8);
    async_ld_b128(&Bs[0][ldRow][ldHalf],     wBase);
    async_ld_b128(&Bs[0][ldRow][ldHalf + 8], wBase + 8);
    wait_async0();
    __syncthreads();

    const int nIter = K >> 5;
    for (int it = 0; it < nIter; ++it) {
        const int buf = it & 1;
        if (it + 1 < nIter) {
            const __bf16* aN = aBase + (size_t)(it + 1) * 32;
            const __bf16* wN = wBase + (size_t)(it + 1) * 32;
            async_ld_b128(&As[buf ^ 1][ldRow][ldHalf],     aN);
            async_ld_b128(&As[buf ^ 1][ldRow][ldHalf + 8], aN + 8);
            async_ld_b128(&Bs[buf ^ 1][ldRow][ldHalf],     wN);
            async_ld_b128(&Bs[buf ^ 1][ldRow][ldHalf + 8], wN + 8);
            if (it + 2 < nIter) {            // emits global_prefetch_b8
                __builtin_prefetch(aN + 32, 0, 0);
                __builtin_prefetch(wN + 32, 0, 0);
            }
        }

        const int mBase = waveRow * 64;
        const int nBase = waveCol * 32;
        const int mr = lane & 15;
        const int kA = (lane < 16) ? 0 : 8;    // A 16x32 bf16 lane layout
        const int kB = (lane < 16) ? 0 : 8;    // B 32x16 bf16 lane layout

        v16bf af[4];
        #pragma unroll
        for (int tb = 0; tb < 4; ++tb) {
            int m = mBase + tb * 16 + mr;
            #pragma unroll
            for (int e = 0; e < 8; ++e) {
                af[tb][e]     = As[buf][m][kA + e];
                af[tb][e + 8] = As[buf][m][kA + 16 + e];
            }
        }
        v16bf bfv[2];
        #pragma unroll
        for (int cbI = 0; cbI < 2; ++cbI) {
            int n = nBase + cbI * 16 + mr;
            #pragma unroll
            for (int e = 0; e < 8; ++e) {
                bfv[cbI][e]     = Bs[buf][n][kB + e];
                bfv[cbI][e + 8] = Bs[buf][n][kB + 16 + e];
            }
        }
        #pragma unroll
        for (int tb = 0; tb < 4; ++tb)
            #pragma unroll
            for (int cbI = 0; cbI < 2; ++cbI)
                acc[tb][cbI] = __builtin_amdgcn_wmma_f32_16x16x32_bf16(
                    false, af[tb], false, bfv[cbI], (short)0, acc[tb][cbI], false, false);

        wait_async0();      // own async fills landed (in-order per wave)
        __syncthreads();    // publish all waves' fills before next compute
    }

    // C layout: VGPR i -> rows i (lanes 0-15) and i+8 (lanes 16-31), N = lane%16
    const int nOff = lane & 15;
    const int rAdd = (lane < 16) ? 0 : 8;
    #pragma unroll
    for (int tb = 0; tb < 4; ++tb)
        #pragma unroll
        for (int cbI = 0; cbI < 2; ++cbI) {
            int col = colG0 + waveCol * 32 + cbI * 16 + nOff;
            float bv = bias ? bias[col] : 0.f;
            #pragma unroll
            for (int i = 0; i < 8; ++i) {
                int row = rowG0 + waveRow * 64 + tb * 16 + rAdd + i;
                C[(size_t)row * M + col] = acc[tb][cbI][i] + bv;
            }
        }
}

// --------------------------------------------------------- rmsnorm (+residual)
__global__ __launch_bounds__(256)
void k_addnorm(const float* __restrict__ X, const float* __restrict__ R,
               const float* __restrict__ w, float* __restrict__ out) {
    __shared__ float red[256];
    int row = blockIdx.x, tid = threadIdx.x;
    size_t base = (size_t)row * Dm;
    float a0 = X[base + tid]       + (R ? R[base + tid]       : 0.f);
    float a1 = X[base + tid + 256] + (R ? R[base + tid + 256] : 0.f);
    float ss = blockReduceSum(a0 * a0 + a1 * a1, red);
    float r = rsqrtf(ss / (float)Dm + 1e-6f);
    out[base + tid]       = a0 * r * w[tid];
    out[base + tid + 256] = a1 * r * w[tid + 256];
}

__global__ void k_resadd(float* __restrict__ y, const float* __restrict__ x, size_t n) {
    for (size_t i = (size_t)blockIdx.x * 256 + threadIdx.x; i < n; i += (size_t)gridDim.x * 256)
        y[i] += x[i];
}

__global__ void k_silu_mul(const float* __restrict__ g, const float* __restrict__ u,
                           __bf16* __restrict__ y, size_t n) {
    for (size_t i = (size_t)blockIdx.x * 256 + threadIdx.x; i < n; i += (size_t)gridDim.x * 256) {
        float gv = g[i];
        float s = gv / (1.f + expf(-gv));
        y[i] = (__bf16)(s * u[i]);
    }
}

// ----------------------------------------- entropy-weighted patch pooling
__global__ __launch_bounds__(256)
void k_patch_pool(const float* __restrict__ X, const float* __restrict__ ENT,
                  __bf16* __restrict__ out) {
    int bp = blockIdx.x;
    int p = bp % NPp, b = bp / NPp;
    int tid = threadIdx.x;
    size_t ebase = (size_t)b * Tb + (size_t)p * PATCHs;
    float e[PATCHs], m = -1e30f;
    #pragma unroll
    for (int j = 0; j < PATCHs; ++j) { e[j] = ENT[ebase + j]; m = fmaxf(m, e[j]); }
    float s = 0.f;
    #pragma unroll
    for (int j = 0; j < PATCHs; ++j) { e[j] = expf(e[j] - m); s += e[j]; }
    float inv = 1.f / s;
    size_t xbase = ebase * Dm;
    for (int d = tid; d < Dm; d += 256) {
        float acc = 0.f;
        #pragma unroll
        for (int j = 0; j < PATCHs; ++j) acc += e[j] * X[xbase + (size_t)j * Dm + d];
        out[(size_t)bp * Dm + d] = (__bf16)(acc * inv);
    }
}

// --------------------------------------- dilated sparse attention (A<=13, dh=64)
__global__ __launch_bounds__(256)
void k_sparse_attn(const float* __restrict__ Q, const float* __restrict__ K,
                   const float* __restrict__ V, float* __restrict__ O, int Tt) {
    int row = blockIdx.x;              // b*Tt + t
    int t = row % Tt;
    int rowBase = row - t;
    int h = threadIdx.x >> 5;          // 8 heads = 8 waves
    int lane = threadIdx.x & 31;
    const float scale = 0.125f;        // dh^-0.5, dh=64

    int pos[16]; int A = 0;
    pos[A++] = t;
    if (t >= 1) pos[A++] = t - 1;
    if (t >= 2) pos[A++] = t - 2;
    for (int k2 = 4; k2 <= t; k2 <<= 1) pos[A++] = t - k2;

    size_t qoff = (size_t)row * Dm + h * 64;
    float q0 = Q[qoff + lane], q1 = Q[qoff + lane + 32];
    float sc[16];
    for (int a = 0; a < A; ++a) {
        size_t koff = (size_t)(rowBase + pos[a]) * Dm + h * 64;
        float p = q0 * K[koff + lane] + q1 * K[koff + lane + 32];
        p += __shfl_xor(p, 16);
        p += __shfl_xor(p, 8);
        p += __shfl_xor(p, 4);
        p += __shfl_xor(p, 2);
        p += __shfl_xor(p, 1);
        sc[a] = p * scale;
    }
    float m = -1e30f;
    for (int a = 0; a < A; ++a) m = fmaxf(m, sc[a]);
    float s = 0.f;
    for (int a = 0; a < A; ++a) { sc[a] = expf(sc[a] - m); s += sc[a]; }
    float inv = 1.f / s;
    float o0 = 0.f, o1 = 0.f;
    for (int a = 0; a < A; ++a) {
        size_t voff = (size_t)(rowBase + pos[a]) * Dm + h * 64;
        o0 += sc[a] * V[voff + lane];
        o1 += sc[a] * V[voff + lane + 32];
    }
    O[qoff + lane]      = o0 * inv;
    O[qoff + lane + 32] = o1 * inv;
}

// ============================================================== launcher
#define IN(i) ((const float*)d_in[i])

extern "C" void kernel_launch(void* const* d_in, const int* in_sizes, int n_in,
                              void* d_out, int out_size, void* d_ws, size_t ws_size,
                              hipStream_t stream) {
    (void)in_sizes; (void)n_in; (void)out_size; (void)ws_size;
    char* ws = (char*)d_ws;
    constexpr size_t MB = 1ull << 20;
    // byte-level arena
    float*  EMB   = (float*)(ws + 0);          // 64MB  (B*T,D) f32
    float*  ENT   = (float*)(ws + 64 * MB);    // 128KB
    float*  GATED = (float*)(ws + 65 * MB);    // 64MB (reused as pid_out, then patch arena)
    float*  Hh    = (float*)(ws + 129 * MB);   // 64MB
    float*  INTEG = (float*)(ws + 193 * MB);   // 64MB
    __bf16* ACAT  = (__bf16*)(ws + 257 * MB);  // 96MB (32768 x 1536 bf16)
    __bf16* WPK   = (__bf16*)(ws + 353 * MB);  // 8MB weight pack scratch
    // patch-level arena (overlays GATED..ACAT after byte phase)
    __bf16* APOOL = (__bf16*)(ws + 65 * MB);   // 4MB (4096x512 bf16)
    float*  XP    = (float*)(ws + 70 * MB);    // 8MB
    float*  XN    = (float*)(ws + 78 * MB);    // 8MB
    float*  INTP  = (float*)(ws + 86 * MB);    // 8MB
    __bf16* ACATP = (__bf16*)(ws + 94 * MB);   // 12MB (4096x1536 bf16)
    float*  Qb    = (float*)(ws + 107 * MB);
    float*  Kb    = (float*)(ws + 115 * MB);
    float*  Vb    = (float*)(ws + 123 * MB);
    float*  ATT   = (float*)(ws + 131 * MB);
    float*  TMP   = (float*)(ws + 139 * MB);
    __bf16* ABUF  = (__bf16*)(ws + 147 * MB);  // 16MB (4096x2048 bf16)
    float*  GBUF  = (float*)(ws + 163 * MB);   // 32MB
    float*  UBUF  = (float*)(ws + 195 * MB);   // 32MB

    const int* bytes = (const int*)d_in[0];
    dim3 blk(256);

    // ---- byte-level phase
    k_embed_entropy<<<NTrows, blk, 0, stream>>>(bytes, IN(1), IN(2), IN(3), IN(4),
                                                IN(5), IN(6), EMB, ENT);
    k_lnn_conv<<<NTrows, blk, 0, stream>>>(EMB, IN(7), IN(8), IN(9), IN(10), GATED);
    k_lnn_scan<<<(Bb * Dm + 255) / 256, blk, 0, stream>>>(GATED, IN(11), Hh, INTEG);
    k_pid_prep<<<NTrows, blk, 0, stream>>>(Hh, INTEG, IN(15), IN(16), ACAT, Tb);
    k_pack_w3<<<3072, blk, 0, stream>>>(IN(12), IN(13), IN(14), WPK, Dm, Dm);
    k_gemm_bf16<<<dim3(Dm / 128, NTrows / 128), blk, 0, stream>>>(
        ACAT, WPK, IN(17), GATED, NTrows, 3 * Dm, Dm);
    k_addnorm<<<NTrows, blk, 0, stream>>>(GATED, EMB, IN(18), EMB);  // x_bytes -> EMB

    // ---- dynamic patching
    k_patch_pool<<<NProws, blk, 0, stream>>>(EMB, ENT, APOOL);
    k_f32_to_bf16<<<1024, blk, 0, stream>>>(IN(19), WPK, (size_t)Dm * Dm);
    k_gemm_bf16<<<dim3(Dm / 128, NProws / 128), blk, 0, stream>>>(
        APOOL, WPK, IN(20), XP, NProws, Dm, Dm);

    // ---- transformer blocks
    for (int L = 0; L < 4; ++L) {
        int bbi = 21 + L * 25;
        const float *norm1 = IN(bbi + 0);
        const float *qWp = IN(bbi + 1), *qWi = IN(bbi + 2), *qWd = IN(bbi + 3),
                    *qgW = IN(bbi + 4), *qgB = IN(bbi + 5);
        const float *kWp = IN(bbi + 6), *kWi = IN(bbi + 7), *kWd = IN(bbi + 8),
                    *kgW = IN(bbi + 9), *kgB = IN(bbi + 10);
        const float *vWp = IN(bbi + 11), *vWi = IN(bbi + 12), *vWd = IN(bbi + 13),
                    *vgW = IN(bbi + 14), *vgB = IN(bbi + 15);
        const float *oW = IN(bbi + 16), *norm2 = IN(bbi + 17);
        const float *fWp = IN(bbi + 18), *fWi = IN(bbi + 19), *fWd = IN(bbi + 20),
                    *fgW = IN(bbi + 21), *fgB = IN(bbi + 22);
        const float *fUp = IN(bbi + 23), *fDown = IN(bbi + 24);

        // attention
        k_addnorm<<<NProws, blk, 0, stream>>>(XP, nullptr, norm1, XN);
        k_cumsum_mean<<<(Bb * Dm + 255) / 256, blk, 0, stream>>>(XN, INTP, NPp);
        k_pid_prep<<<NProws, blk, 0, stream>>>(XN, INTP, qgW, qgB, ACATP, NPp);
        k_pack_w3<<<3072, blk, 0, stream>>>(qWp, qWi, qWd, WPK, Dm, Dm);
        k_gemm_bf16<<<dim3(Dm / 128, NProws / 128), blk, 0, stream>>>(
            ACATP, WPK, nullptr, Qb, NProws, 3 * Dm, Dm);
        k_pid_prep<<<NProws, blk, 0, stream>>>(XN, INTP, kgW, kgB, ACATP, NPp);
        k_pack_w3<<<3072, blk, 0, stream>>>(kWp, kWi, kWd, WPK, Dm, Dm);
        k_gemm_bf16<<<dim3(Dm / 128, NProws / 128), blk, 0, stream>>>(
            ACATP, WPK, nullptr, Kb, NProws, 3 * Dm, Dm);
        k_pid_prep<<<NProws, blk, 0, stream>>>(XN, INTP, vgW, vgB, ACATP, NPp);
        k_pack_w3<<<3072, blk, 0, stream>>>(vWp, vWi, vWd, WPK, Dm, Dm);
        k_gemm_bf16<<<dim3(Dm / 128, NProws / 128), blk, 0, stream>>>(
            ACATP, WPK, nullptr, Vb, NProws, 3 * Dm, Dm);
        k_sparse_attn<<<NProws, blk, 0, stream>>>(Qb, Kb, Vb, ATT, NPp);
        k_f32_to_bf16<<<8192, blk, 0, stream>>>(ATT, APOOL, (size_t)NProws * Dm);
        k_f32_to_bf16<<<1024, blk, 0, stream>>>(oW, WPK, (size_t)Dm * Dm);
        k_gemm_bf16<<<dim3(Dm / 128, NProws / 128), blk, 0, stream>>>(
            APOOL, WPK, nullptr, TMP, NProws, Dm, Dm);
        k_resadd<<<8192, blk, 0, stream>>>(XP, TMP, (size_t)NProws * Dm);

        // FFN
        k_addnorm<<<NProws, blk, 0, stream>>>(XP, nullptr, norm2, XN);
        k_cumsum_mean<<<(Bb * Dm + 255) / 256, blk, 0, stream>>>(XN, INTP, NPp);
        k_pid_prep<<<NProws, blk, 0, stream>>>(XN, INTP, fgW, fgB, ACATP, NPp);
        k_pack_w3<<<12288, blk, 0, stream>>>(fWp, fWi, fWd, WPK, FFd, Dm);
        k_gemm_bf16<<<dim3(FFd / 128, NProws / 128), blk, 0, stream>>>(
            ACATP, WPK, nullptr, GBUF, NProws, 3 * Dm, FFd);
        k_f32_to_bf16<<<8192, blk, 0, stream>>>(XN, APOOL, (size_t)NProws * Dm);
        k_f32_to_bf16<<<4096, blk, 0, stream>>>(fUp, WPK, (size_t)FFd * Dm);
        k_gemm_bf16<<<dim3(FFd / 128, NProws / 128), blk, 0, stream>>>(
            APOOL, WPK, nullptr, UBUF, NProws, Dm, FFd);
        k_silu_mul<<<32768, blk, 0, stream>>>(GBUF, UBUF, ABUF, (size_t)NProws * FFd);
        k_f32_to_bf16<<<4096, blk, 0, stream>>>(fDown, WPK, (size_t)Dm * FFd);
        k_gemm_bf16<<<dim3(Dm / 128, NProws / 128), blk, 0, stream>>>(
            ABUF, WPK, nullptr, TMP, NProws, FFd, Dm);
        k_resadd<<<8192, blk, 0, stream>>>(XP, TMP, (size_t)NProws * Dm);
    }

    // ---- head
    k_addnorm<<<NProws, blk, 0, stream>>>(XP, nullptr, IN(121), XN);
    k_f32_to_bf16<<<8192, blk, 0, stream>>>(XN, APOOL, (size_t)NProws * Dm);
    k_f32_to_bf16<<<512, blk, 0, stream>>>(IN(122), WPK, (size_t)256 * Dm);
    k_gemm_bf16<<<dim3(256 / 128, NProws / 128), blk, 0, stream>>>(
        APOOL, WPK, nullptr, (float*)d_out, NProws, Dm, 256);
}
